// TimesNet_PFRP_7911329759657
// MI455X (gfx1250) — matrix operand order repaired and checked by reading, not compiled, not run on hardware
//
#include <hip/hip_runtime.h>
#include <hip/hip_bf16.h>
#include <math.h>

// ---- problem constants (from reference) ----
#define B_   256
#define S_   96
#define P_   720
#define K_   32
#define N_   300000
#define DFF_ 128
#define FD_  16

#define CHUNK 1024
#define NC    ((N_ + CHUNK - 1) / CHUNK)   // 293 chunks
#define CAND  (NC * K_)                    // 9376 candidates / row
#define CGK   (S_ + P_)                    // 816
#define CGKP  832                          // pad to 26*32 for WMMA K
#define KT    (CGKP / 32)                  // 26 K-steps
#define SEGS  (P_ * 4 / 16)                // 180 b128 segments per future row

typedef __attribute__((ext_vector_type(16))) _Float16 v16h;
typedef __attribute__((ext_vector_type(8)))  _Float16 v8h;
typedef __attribute__((ext_vector_type(8)))  float    v8f;

union V16H { v16h v; v8h p[2]; _Float16 h[16]; };

__device__ __forceinline__ float gelu_erf(float x) {
  return 0.5f * x * (1.0f + erff(x * 0.70710678118654752440f));
}
__device__ __forceinline__ float sigmoidf_(float x) {
  return 1.0f / (1.0f + expf(-x));
}

// ---------------------------------------------------------------------------
// fp32 -> fp16 conversion of retrieval keys (stays resident in 192MB L2)
// ---------------------------------------------------------------------------
__global__ void cvt_pf_kernel(const float* __restrict__ pf,
                              _Float16* __restrict__ pf16, int n) {
  int i = blockIdx.x * blockDim.x + threadIdx.x;
  if (i < n) pf16[i] = (_Float16)pf[i];
}

// cg_W1 [816,128] -> f16, transposed+padded to [128][832] so each lane's
// WMMA B-fragment is two contiguous 16B loads (k = 8h.., 16+8h..).
__global__ void cvt_w1t_kernel(const float* __restrict__ w1,
                               _Float16* __restrict__ w1t) {
  int i = blockIdx.x * blockDim.x + threadIdx.x;
  if (i < DFF_ * CGKP) {
    int c = i / CGKP, k = i % CGKP;
    w1t[i] = (k < CGK) ? (_Float16)w1[(size_t)k * DFF_ + c] : (_Float16)0.0f;
  }
}

// ---------------------------------------------------------------------------
// per-row: mean / unbiased std / normalize, frozen encoder MLP, L2-normalize
// ---------------------------------------------------------------------------
__global__ __launch_bounds__(128) void prep_kernel(
    const float* __restrict__ x_enc,
    const float* __restrict__ enc_W1, const float* __restrict__ enc_b1,
    const float* __restrict__ enc_W2, const float* __restrict__ enc_b2,
    float* __restrict__ xn, float* __restrict__ meanv, float* __restrict__ stdv,
    _Float16* __restrict__ feat16) {
  __shared__ float red[128];
  __shared__ float xrow[S_];
  __shared__ float h1[64];
  __shared__ float fv[FD_];
  __shared__ float nrm;
  const int b = blockIdx.x, t = threadIdx.x;
  float x = (t < S_) ? x_enc[(size_t)b * S_ + t] : 0.0f;
  red[t] = x; __syncthreads();
  for (int off = 64; off > 0; off >>= 1) {
    if (t < off) red[t] += red[t + off];
    __syncthreads();
  }
  const float mean = red[0] / (float)S_;
  __syncthreads();
  float d = (t < S_) ? (x - mean) : 0.0f;
  red[t] = d * d; __syncthreads();
  for (int off = 64; off > 0; off >>= 1) {
    if (t < off) red[t] += red[t + off];
    __syncthreads();
  }
  float sd = sqrtf(red[0] / (float)(S_ - 1));   // ddof=1 (torch.std)
  if (sd == 0.0f) sd = 1e-6f;
  if (t < S_) { float v = (x - mean) / sd; xrow[t] = v; xn[(size_t)b * S_ + t] = v; }
  if (t == 0) { meanv[b] = mean; stdv[b] = sd; }
  __syncthreads();
  if (t < 64) {
    float s = enc_b1[t];
    for (int i = 0; i < S_; ++i) s += xrow[i] * enc_W1[i * 64 + t];
    h1[t] = fmaxf(s, 0.0f);
  }
  __syncthreads();
  if (t < FD_) {
    float s = enc_b2[t];
    for (int i = 0; i < 64; ++i) s += h1[i] * enc_W2[i * FD_ + t];
    fv[t] = s;
  }
  __syncthreads();
  if (t == 0) {
    float s = 0.f;
    for (int i = 0; i < FD_; ++i) s += fv[i] * fv[i];
    nrm = fmaxf(sqrtf(s), 1e-12f);
  }
  __syncthreads();
  if (t < FD_) feat16[(size_t)b * FD_ + t] = (_Float16)(fv[t] / nrm);
}

// ---------------------------------------------------------------------------
// WMMA cosine GEMM (16 batch rows x 1024 key cols per block) + per-chunk
// top-32 selection.  f16 inputs, f32 accumulate, K=16 zero-padded to 32.
// ---------------------------------------------------------------------------
__global__ __launch_bounds__(256) void cos_topk_kernel(
    const _Float16* __restrict__ feat16, const _Float16* __restrict__ pf16,
    float* __restrict__ candVal, int* __restrict__ candIdx) {
  __shared__ float cosLDS[16][CHUNK];           // 64KB
  const int tid  = threadIdx.x;
  const int wave = tid >> 5;
  const int lane = tid & 31;
  const int l16  = lane & 15;
  const bool hi  = lane >= 16;
  const int rowBase   = blockIdx.y * 16;
  const int chunkBase = blockIdx.x * CHUNK;

  // A fragment: 16-bit A 16x32 layout; halves 8..15 are K=16..31 padding (=0)
  V16H a;
  a.p[1] = (v8h){};
  a.p[0] = *(const v8h*)(feat16 + (size_t)(rowBase + l16) * FD_ + (hi ? 8 : 0));

  for (int s = 0; s < CHUNK / (16 * 8); ++s) {  // 8 column-tiles per wave
    const int t   = wave * (CHUNK / (16 * 8)) + s;
    const int n0  = chunkBase + t * 16;
    const int col = n0 + l16;
    V16H bf;
    if (col < N_) {
      // 16B of pf16 row: K = 8*hi + 0..7; halves 8..15 are K>=16 padding
      bf.p[0] = *(const v8h*)(pf16 + (size_t)col * FD_ + (hi ? 8 : 0));
      __builtin_prefetch(pf16 + (size_t)(col + 16) * FD_, 0, 0);
      bf.p[1] = (v8h){};
    } else {
      bf.p[0] = (v8h){};
      bf.p[1] = (v8h){};                         // K>=16 padding rows / OOB
    }
    v8f acc = {};
    acc = __builtin_amdgcn_wmma_f32_16x16x32_f16(
        false, a.v, false, bf.v, (short)0, acc, false, false);
    const int cc = t * 16 + l16;                 // D: lane -> N
    const bool valid = (n0 + l16) < N_;
#pragma unroll
    for (int r = 0; r < 8; ++r) {                // D: VGPR r -> M (+8 hi half)
      const int m = r + (hi ? 8 : 0);
      cosLDS[m][cc] = valid ? acc[r] : -3.0e38f;
    }
  }
  __syncthreads();

  // per-wave iterative argmax: wave owns rows 2w, 2w+1
  for (int rr = wave * 2; rr < wave * 2 + 2; ++rr) {
    const int grow = rowBase + rr;
    for (int it = 0; it < K_; ++it) {
      float bv = -3.0e38f; int bi = 0;
      for (int c = lane; c < CHUNK; c += 32) {
        float v = cosLDS[rr][c];
        if (v > bv) { bv = v; bi = c; }
      }
#pragma unroll
      for (int off = 16; off > 0; off >>= 1) {
        float ov = __shfl_xor(bv, off);
        int   oi = __shfl_xor(bi, off);
        if (ov > bv || (ov == bv && oi < bi)) { bv = ov; bi = oi; }
      }
      if (lane == 0) {
        cosLDS[rr][bi] = -3.0e38f;               // knock out winner
        size_t o = ((size_t)grow * NC + blockIdx.x) * K_ + it;
        candVal[o] = bv;
        candIdx[o] = chunkBase + bi;
      }
      __builtin_amdgcn_wave_barrier();           // keep intra-wave DS order
    }
  }
}

// ---------------------------------------------------------------------------
// merge 293x32 per-chunk candidates -> global top-32 (descending, like top_k)
// ---------------------------------------------------------------------------
__global__ __launch_bounds__(256) void merge_topk_kernel(
    float* __restrict__ candVal, const int* __restrict__ candIdx,
    float* __restrict__ topkVal, int* __restrict__ topkIdx) {
  __shared__ float rv[256];
  __shared__ int   ri[256];
  const int b = blockIdx.x, t = threadIdx.x;
  float* cv = candVal + (size_t)b * CAND;
  const int* ci = candIdx + (size_t)b * CAND;
  for (int it = 0; it < K_; ++it) {
    float bv = -3.0e38f; int bi = 0;
    for (int c = t; c < CAND; c += 256) {
      float v = cv[c];
      if (v > bv) { bv = v; bi = c; }
    }
    rv[t] = bv; ri[t] = bi;
    __syncthreads();
    for (int off = 128; off > 0; off >>= 1) {
      if (t < off) {
        if (rv[t + off] > rv[t] ||
            (rv[t + off] == rv[t] && ri[t + off] < ri[t])) {
          rv[t] = rv[t + off]; ri[t] = ri[t + off];
        }
      }
      __syncthreads();
    }
    if (t == 0) {
      int c = ri[0];
      topkVal[b * K_ + it] = rv[0];
      topkIdx[b * K_ + it] = ci[c];
      cv[c] = -3.0e38f;
    }
    __syncthreads();
  }
}

// ---------------------------------------------------------------------------
// confidence gate: WMMA GEMM [16 rows x 128 cols], K=816 (pad 832).
// Future rows are gathered global->LDS with the gfx1250 async copy engine
// (GLOBAL_LOAD_ASYNC_TO_LDS_B128, ASYNCcnt), overlapping with xn staging;
// then converted to the f16 A-tile.  layer2 (128->1) via ds_add_f32 atomics.
// ---------------------------------------------------------------------------
__global__ __launch_bounds__(256) void conf_kernel(
    const float* __restrict__ xn, const float* __restrict__ future,
    const int* __restrict__ topkIdx, const float* __restrict__ topkVal,
    const _Float16* __restrict__ w1t, const float* __restrict__ cg_b1,
    const float* __restrict__ cg_W2, const float* __restrict__ cg_b2,
    float* __restrict__ sim) {
  __shared__ alignas(16) float    stage[16][P_];  // 46.1KB fp32 gather target
  __shared__ alignas(32) _Float16 aT[16][CGKP];   // 26.6KB f16 WMMA A-tile
  __shared__ float accM[16];
  const int tid = threadIdx.x;
  const int rowBase = blockIdx.x * 16;            // flattened (b*K + k)

  // 1) async gather: 16 rows x 180 x b128 = 46KB global->LDS, no VGPR hop
  for (int i = tid; i < 16 * SEGS; i += 256) {
    const int r = i / SEGS, seg = i % SEGS;
    const int grow = rowBase + r;
    const int b = grow >> 5, k = grow & 31;
    const float* gsrc =
        future + (size_t)topkIdx[b * K_ + k] * P_ + (size_t)seg * 4;
    unsigned lds_off = (unsigned)(size_t)&stage[r][seg * 4];
    asm volatile("global_load_async_to_lds_b128 %0, %1, off"
                 :: "v"(lds_off), "v"(gsrc)
                 : "memory");
  }

  // 2) overlap: stage xn part + K padding of the A-tile while copies fly
  for (int i = tid; i < 16 * CGKP; i += 256) {
    const int r = i / CGKP, c = i % CGKP;
    if (c < S_) {
      const int b = (rowBase + r) >> 5;
      aT[r][c] = (_Float16)xn[(size_t)b * S_ + c];
    } else if (c >= CGK) {
      aT[r][c] = (_Float16)0.0f;
    }
  }
  if (tid < 16) accM[tid] = cg_b2[0];

  asm volatile("s_wait_asynccnt 0x0" ::: "memory");
  __syncthreads();

  // 3) fp32 -> fp16 conversion of the gathered future rows
  for (int i = tid; i < 16 * P_; i += 256) {
    const int r = i / P_, c = i % P_;
    aT[r][S_ + c] = (_Float16)stage[r][c];
  }
  __syncthreads();

  const int wave = tid >> 5, lane = tid & 31, l16 = lane & 15;
  const bool hi  = lane >= 16;
  const int col  = wave * 16 + l16;               // output column (0..127)
  v8f acc = {};
  for (int kt = 0; kt < KT; ++kt) {
    const int kb = kt * 32;
    V16H a, bw;
    a.p[0] = *(const v8h*)(&aT[l16][kb + (hi ? 8 : 0)]);
    a.p[1] = *(const v8h*)(&aT[l16][kb + 16 + (hi ? 8 : 0)]);
    bw.p[0] = *(const v8h*)(w1t + (size_t)col * CGKP + kb + (hi ? 8 : 0));
    bw.p[1] = *(const v8h*)(w1t + (size_t)col * CGKP + kb + 16 + (hi ? 8 : 0));
    acc = __builtin_amdgcn_wmma_f32_16x16x32_f16(
        false, a.v, false, bw.v, (short)0, acc, false, false);
  }
  const float bias = cg_b1[col];
  const float w2c  = cg_W2[col];
#pragma unroll
  for (int r = 0; r < 8; ++r) {
    const int m = r + (hi ? 8 : 0);
    atomicAdd(&accM[m], gelu_erf(acc[r] + bias) * w2c);  // ds_add_f32
  }
  __syncthreads();
  if (tid < 16) {
    const int grow = rowBase + tid;
    const int b = grow >> 5, k = grow & 31;
    sim[grow] = topkVal[b * K_ + k] * sigmoidf_(accM[tid]);
  }
}

// ---------------------------------------------------------------------------
// per-row tail: softmax(sim), weight MLP, output-gate MLP, fusion, blend
// ---------------------------------------------------------------------------
__global__ __launch_bounds__(256) void final_kernel(
    const float* __restrict__ sim, const float* __restrict__ xn,
    const float* __restrict__ meanv, const float* __restrict__ stdv,
    const int* __restrict__ topkIdx, const float* __restrict__ future,
    const float* __restrict__ og_W1, const float* __restrict__ og_b1,
    const float* __restrict__ og_W2, const float* __restrict__ og_b2,
    const float* __restrict__ wm_W1, const float* __restrict__ wm_b1,
    const float* __restrict__ wm_W2, const float* __restrict__ wm_b2,
    const float* __restrict__ y1, float* __restrict__ out) {
  __shared__ float simL[K_];
  __shared__ float wk[K_];
  __shared__ float hwm[DFF_];
  __shared__ float h2[2 * S_];
  __shared__ float wt[2];
  const int b = blockIdx.x, t = threadIdx.x;
  if (t < K_) simL[t] = sim[b * K_ + t];
  __syncthreads();
  if (t < DFF_) {
    float s = wm_b1[t];
    for (int i = 0; i < K_; ++i) s += simL[i] * wm_W1[i * DFF_ + t];
    hwm[t] = gelu_erf(s);
  }
  if (t < 2 * S_) {
    float s = og_b1[t];
    for (int i = 0; i < S_; ++i) s += xn[(size_t)b * S_ + i] * og_W1[i * (2 * S_) + t];
    h2[t] = gelu_erf(s);
  }
  __syncthreads();
  if (t == 0) {
    float m = simL[0];
    for (int i = 1; i < K_; ++i) m = fmaxf(m, simL[i]);
    float ssum = 0.f;
    for (int i = 0; i < K_; ++i) { float e = expf(simL[i] - m); wk[i] = e; ssum += e; }
    for (int i = 0; i < K_; ++i) wk[i] /= ssum;
    float o0 = wm_b2[0], o1 = wm_b2[1];
    for (int i = 0; i < DFF_; ++i) {
      o0 += hwm[i] * wm_W2[i * 2 + 0];
      o1 += hwm[i] * wm_W2[i * 2 + 1];
    }
    float mm = fmaxf(o0, o1);
    float e0 = expf(o0 - mm), e1 = expf(o1 - mm);
    wt[0] = e0 / (e0 + e1); wt[1] = e1 / (e0 + e1);
  }
  __syncthreads();
  const float mean = meanv[b], sd = stdv[b];
  for (int p = t; p < P_; p += 256) {
    float fus = 0.f;
    for (int k = 0; k < K_; ++k)
      fus += wk[k] * future[(size_t)topkIdx[b * K_ + k] * P_ + p];
    float og0 = og_b2[p], og1 = og_b2[P_ + p];
    for (int j = 0; j < 2 * S_; ++j) {
      const float hj = h2[j];
      og0 += hj * og_W2[(size_t)j * (2 * P_) + p];
      og1 += hj * og_W2[(size_t)j * (2 * P_) + P_ + p];
    }
    float y = (fus * og0 + og1) * sd + mean;
    out[(size_t)b * P_ + p] = y * wt[0] + y1[(size_t)b * P_ + p] * wt[1];
  }
}

// ---------------------------------------------------------------------------
extern "C" void kernel_launch(void* const* d_in, const int* in_sizes, int n_in,
                              void* d_out, int out_size, void* d_ws, size_t ws_size,
                              hipStream_t stream) {
  const float* x_enc  = (const float*)d_in[0];
  const float* y1     = (const float*)d_in[1];
  const float* enc_W1 = (const float*)d_in[2];
  const float* enc_b1 = (const float*)d_in[3];
  const float* enc_W2 = (const float*)d_in[4];
  const float* enc_b2 = (const float*)d_in[5];
  const float* cg_W1  = (const float*)d_in[6];
  const float* cg_b1  = (const float*)d_in[7];
  const float* cg_W2  = (const float*)d_in[8];
  const float* cg_b2  = (const float*)d_in[9];
  const float* og_W1  = (const float*)d_in[10];
  const float* og_b1  = (const float*)d_in[11];
  const float* og_W2  = (const float*)d_in[12];
  const float* og_b2  = (const float*)d_in[13];
  const float* wm_W1  = (const float*)d_in[14];
  const float* wm_b1  = (const float*)d_in[15];
  const float* wm_W2  = (const float*)d_in[16];
  const float* wm_b2  = (const float*)d_in[17];
  const float* pf     = (const float*)d_in[18];
  const float* fut    = (const float*)d_in[19];
  float* out = (float*)d_out;
  (void)in_sizes; (void)n_in; (void)out_size; (void)ws_size;

  char* ws = (char*)d_ws;
  size_t off = 0;
  auto alloc = [&](size_t bytes) {
    char* p = ws + off;
    off = (off + bytes + 255) & ~(size_t)255;
    return p;
  };
  _Float16* pf16    = (_Float16*)alloc((size_t)N_ * FD_ * 2);
  _Float16* w1t     = (_Float16*)alloc((size_t)DFF_ * CGKP * 2);
  _Float16* feat16  = (_Float16*)alloc((size_t)B_ * FD_ * 2);
  float*    xn      = (float*)alloc((size_t)B_ * S_ * 4);
  float*    meanv   = (float*)alloc((size_t)B_ * 4);
  float*    stdv    = (float*)alloc((size_t)B_ * 4);
  float*    candVal = (float*)alloc((size_t)B_ * CAND * 4);
  int*      candIdx = (int*)alloc((size_t)B_ * CAND * 4);
  float*    topkVal = (float*)alloc((size_t)B_ * K_ * 4);
  int*      topkIdx = (int*)alloc((size_t)B_ * K_ * 4);
  float*    sim     = (float*)alloc((size_t)B_ * K_ * 4);

  cvt_pf_kernel<<<(N_ * FD_ + 255) / 256, 256, 0, stream>>>(pf, pf16, N_ * FD_);
  cvt_w1t_kernel<<<(DFF_ * CGKP + 255) / 256, 256, 0, stream>>>(cg_W1, w1t);
  prep_kernel<<<B_, 128, 0, stream>>>(x_enc, enc_W1, enc_b1, enc_W2, enc_b2,
                                      xn, meanv, stdv, feat16);
  cos_topk_kernel<<<dim3(NC, B_ / 16), 256, 0, stream>>>(feat16, pf16,
                                                         candVal, candIdx);
  merge_topk_kernel<<<B_, 256, 0, stream>>>(candVal, candIdx, topkVal, topkIdx);
  conf_kernel<<<(B_ * K_) / 16, 256, 0, stream>>>(xn, fut, topkIdx, topkVal,
                                                  w1t, cg_b1, cg_W2, cg_b2, sim);
  final_kernel<<<B_, 256, 0, stream>>>(sim, xn, meanv, stdv, topkIdx, fut,
                                       og_W1, og_b1, og_W2, og_b2,
                                       wm_W1, wm_b1, wm_W2, wm_b2, y1, out);
}
